// EquivariantHardAlignmentModel_45243185496757
// MI455X (gfx1250) — matrix operand order repaired and checked by reading, not compiled, not run on hardware
//
#include <hip/hip_runtime.h>
#include <hip/hip_bf16.h>

// ---- problem constants (from reference) ----
#define B_SZ   32
#define N_SEQ  512          // N_ENC == N_DEC
#define IN_V   2000
#define OUT_V  2000
#define H_DIM  256
#define F_DIM  256
#define K_TAPS 5
#define P_G    4
#define E_DIM  128
#define H4     1024         // 4*H

typedef __attribute__((ext_vector_type(16))) _Float16 v16h;
typedef __attribute__((ext_vector_type(8)))  float    v8f;

__device__ __forceinline__ int lane_id() { return threadIdx.x & 31; }

// A fragment (16x32 f16), A row-major, base pre-offset to (m0,k0), row stride lda.
// Contiguous-K order used here (frag_bt matches): lane l -> row m0+(l&15); lanes 0-15 hold K 0..15; lanes 16-31 hold K 16..31.
__device__ __forceinline__ v16h frag_a(const _Float16* A, int lda) {
  int l = lane_id();
  const _Float16* p = A + (size_t)(l & 15) * lda + ((l >> 4) << 4);
  v16h a;
#pragma unroll
  for (int e = 0; e < 16; ++e) a[e] = p[e];
  return a;
}

// B fragment (32x16 f16) where storage is N-major: W[n][k], row stride ldw (logical B = W^T).
// Base pre-offset to (n0,k0). Contiguous 16-half read per lane -> 2x global_load_b128.
__device__ __forceinline__ v16h frag_bt(const _Float16* W, int ldw) {
  int l = lane_id();
  const _Float16* p = W + (size_t)(l & 15) * ldw + ((l >> 4) << 4);
  v16h b;
#pragma unroll
  for (int e = 0; e < 16; ++e) b[e] = p[e];
  return b;
}

__device__ __forceinline__ v8f wmma_f16(v16h a, v16h b, v8f c) {
  return __builtin_amdgcn_wmma_f32_16x16x32_f16(false, a, false, b, (short)0, c, false, false);
}

__device__ __forceinline__ float sigmoidf_(float x) { return 1.f / (1.f + expf(-x)); }

// -------------------- small utility kernels --------------------
__global__ void k_init_out(float* out) {
  if (blockIdx.x == 0 && threadIdx.x == 0) out[0] = 0.f;
}

__global__ void k_f32_to_f16(const float* __restrict__ s, _Float16* __restrict__ d, int n) {
  int i = blockIdx.x * blockDim.x + threadIdx.x;
  if (i < n) d[i] = (_Float16)s[i];
}

// dst[bat][c][r] = (f16) src[bat][r][c]  — makes every WMMA B operand N-major (contiguous K)
__global__ void k_transpose_to_f16(const float* __restrict__ src, _Float16* __restrict__ dst,
                                   int R, int C, int n) {
  int i = blockIdx.x * blockDim.x + threadIdx.x;
  if (i >= n) return;
  int bat = i / (R * C);
  int rem = i - bat * R * C;
  int c = rem / R;
  int r = rem - c * R;
  dst[i] = (_Float16)src[(size_t)bat * R * C + (size_t)r * C + c];
}

// e_pad[b][r][f] = r in [2, N+2) ? tanh(gembed[xs[b][r-2]][f]) : 0   (SAME-conv halo rows)
__global__ void k_embed_tanh_pad(const int* __restrict__ xs, const float* __restrict__ W,
                                 _Float16* __restrict__ epad) {
  const int nrow = N_SEQ + 4;
  int i = blockIdx.x * blockDim.x + threadIdx.x;
  if (i >= B_SZ * nrow * F_DIM) return;
  int f = i % F_DIM;
  int r = (i / F_DIM) % nrow;
  int b = i / (F_DIM * nrow);
  float v = 0.f;
  int pos = r - 2;
  if (pos >= 0 && pos < N_SEQ) {
    int id = xs[b * N_SEQ + pos];
    v = tanhf(W[(size_t)id * F_DIM + f]);
  }
  epad[i] = (_Float16)v;
}

// -------------------- conv1d as WMMA GEMM: aconv = tanh(conv(e)) --------------------
// WcT layout: [k][fout][fin] (transposed) so B fragments are contiguous-K loads.
__global__ void __launch_bounds__(256)
k_conv_gemm(const _Float16* __restrict__ epad, const _Float16* __restrict__ WcT,
            _Float16* __restrict__ aconv) {
  int wave = (blockIdx.x * blockDim.x + threadIdx.x) >> 5;
  int mt = wave >> 4;            // 1024 M-tiles (B*N/16)
  int nt = wave & 15;            // 16 N-tiles (F/16)
  int p0 = mt * 16;              // global position b*N + i  (tile never straddles a batch)
  int b  = p0 / N_SEQ;
  int i0 = p0 % N_SEQ;
  int n0 = nt * 16;
  v8f acc = {};
  for (int k = 0; k < K_TAPS; ++k) {
    const _Float16* Arow = epad + ((size_t)b * (N_SEQ + 4) + (i0 + k)) * F_DIM;
    const _Float16* Bw   = WcT + (size_t)k * F_DIM * F_DIM + (size_t)n0 * F_DIM;
#pragma unroll
    for (int kc = 0; kc < F_DIM / 32; ++kc) {
      v16h a = frag_a(Arow + kc * 32, F_DIM);
      v16h f = frag_bt(Bw + kc * 32, F_DIM);
      acc = wmma_f16(a, f, acc);
    }
  }
  int l = lane_id(), cn = n0 + (l & 15), rb = (l >> 4) << 3;
#pragma unroll
  for (int r = 0; r < 8; ++r)
    aconv[(size_t)(p0 + rb + r) * F_DIM + cn] = (_Float16)tanhf(acc[r]);
}

// -------------------- decode GEMM + softmax(V) + gather(ys) fused --------------------
// one workgroup = 16 rows; 16x2000 f32 logits live in LDS (128000 B < 320 KB/WGP).
// A tile (8 KB, contiguous) staged with CDNA5 async-DMA to LDS (ASYNCcnt path).
// WdT layout: [v][fin] (transposed decode weights).
__global__ void __launch_bounds__(256)
k_decode_softmax_gather(const _Float16* __restrict__ aconv, const _Float16* __restrict__ WdT,
                        const int* __restrict__ ys, float* __restrict__ scores) {
  extern __shared__ char smem_raw[];
  _Float16* As = (_Float16*)smem_raw;                                  // 16 x 256 (at LDS offset 0)
  float*    z  = (float*)(smem_raw + 16 * F_DIM * sizeof(_Float16));   // 16 x 2000
  int p0 = blockIdx.x * 16;
  int b  = p0 / N_SEQ;
  int i0 = p0 % N_SEQ;
  {
    // 256 lanes x 32B = 8 KB: global_load_async_to_lds_b128 (cache/mem -> LDS, no VGPR data)
    unsigned ldsoff = (unsigned)threadIdx.x * 32u;                     // As is at LDS base
    const _Float16* g = aconv + (size_t)p0 * F_DIM + (size_t)threadIdx.x * 16;
    asm volatile("global_load_async_to_lds_b128 %0, %1, off"
                 :: "v"(ldsoff), "v"(g)
                 : "memory");
    asm volatile("s_wait_asynccnt 0x0" ::: "memory");
  }
  __syncthreads();
  int w = threadIdx.x >> 5, l = lane_id();
  for (int nt = w; nt < OUT_V / 16; nt += 8) {     // uniform per wave
    v8f acc = {};
#pragma unroll
    for (int kc = 0; kc < F_DIM / 32; ++kc) {
      v16h a = frag_a(As + kc * 32, F_DIM);
      v16h f = frag_bt(WdT + (size_t)(nt * 16) * F_DIM + kc * 32, F_DIM);
      acc = wmma_f16(a, f, acc);
    }
    int cn = nt * 16 + (l & 15), rb = (l >> 4) << 3;
#pragma unroll
    for (int r = 0; r < 8; ++r) z[(rb + r) * OUT_V + cn] = acc[r];
  }
  __syncthreads();
  for (int m = w; m < 16; m += 8) {                // each wave owns rows {w, w+8}
    float mx = -1e30f;
    for (int n = l; n < OUT_V; n += 32) mx = fmaxf(mx, z[m * OUT_V + n]);
#pragma unroll
    for (int s = 16; s > 0; s >>= 1) mx = fmaxf(mx, __shfl_xor(mx, s, 32));
    float sum = 0.f;
    for (int n = l; n < OUT_V; n += 32) {
      float e = expf(z[m * OUT_V + n] - mx);
      z[m * OUT_V + n] = e;
      sum += e;
    }
#pragma unroll
    for (int s = 16; s > 0; s >>= 1) sum += __shfl_xor(sum, s, 32);
    float inv = 1.f / sum;
    int i = i0 + m;
    for (int j = l; j < N_SEQ; j += 32) {
      int v = ys[b * N_SEQ + j];
      scores[((size_t)(b * N_SEQ + i)) * N_SEQ + j] = z[m * OUT_V + v] * inv;
    }
  }
}

// -------------------- LSTM input projection: pre[t][b][4H] = embed @ Wih^T + bias --------------------
// embedding gather (with <P masking) fused into the A-fragment load; m = t*32+b
__global__ void __launch_bounds__(256)
k_pre_gemm(const int* __restrict__ idx, const float* __restrict__ emb,
           const _Float16* __restrict__ Wih, const float* __restrict__ bias,
           _Float16* __restrict__ pre) {
  int wave = (blockIdx.x * blockDim.x + threadIdx.x) >> 5;
  int mt = wave >> 6;                 // 1024 M-tiles (T*B/16)
  int nt = wave & 63;                 // 64 N-tiles (4H/16)
  int m0 = mt * 16, n0 = nt * 16;
  int l = lane_id();
  int m = m0 + (l & 15);
  int t = m >> 5, b = m & 31;
  int id = idx[b * N_SEQ + t];
  if (id < P_G) id = 0;
  const float* arow = emb + (size_t)id * E_DIM;
  v8f acc = {};
#pragma unroll
  for (int kc = 0; kc < E_DIM / 32; ++kc) {
    int kb = kc * 32 + ((l >> 4) << 4);
    v16h a;
#pragma unroll
    for (int e = 0; e < 16; ++e) a[e] = (_Float16)arow[kb + e];
    v16h f = frag_bt(Wih + (size_t)n0 * E_DIM + kc * 32, E_DIM);
    acc = wmma_f16(a, f, acc);
  }
  int cn = n0 + (l & 15), rb = (l >> 4) << 3;
#pragma unroll
  for (int r = 0; r < 8; ++r)
    pre[(size_t)(m0 + rb + r) * H4 + cn] = (_Float16)(acc[r] + bias[cn]);
}

// -------------------- persistent recurrent LSTM --------------------
// one workgroup (32 waves) per direction; h(f16), c(f32), z(f32 32x1024) resident in LDS.
// per step: 128 16x16 tiles (4/wave) x 8 K-chunks = 1024 WMMA, then fused gate pointwise.
__global__ void __launch_bounds__(1024)
k_lstm(const _Float16* __restrict__ pre, const _Float16* __restrict__ Whh,
       _Float16* __restrict__ hout_fwd, _Float16* __restrict__ hout_bwd,
       const float* __restrict__ h0, const float* __restrict__ c0,
       float* __restrict__ hT, float* __restrict__ cT) {
  int dir = blockIdx.x;                       // 0 = forward in time, 1 = backward
  _Float16* hout = dir ? hout_bwd : hout_fwd;
  extern __shared__ char smem_raw[];
  _Float16* h_lds = (_Float16*)smem_raw;                                      // 32x256 f16
  float*    c_lds = (float*)(smem_raw + B_SZ * H_DIM * sizeof(_Float16));     // 32x256 f32
  float*    z_lds = (float*)((char*)c_lds + B_SZ * H_DIM * sizeof(float));    // 32x1024 f32
  int tid = threadIdx.x, w = tid >> 5, l = tid & 31;

  for (int i = tid; i < B_SZ * H_DIM; i += 1024) {
    h_lds[i] = (_Float16)(h0 ? h0[i] : 0.f);
    c_lds[i] = c0 ? c0[i] : 0.f;
  }
  __syncthreads();

  for (int step = 0; step < N_SEQ; ++step) {
    int tt = dir ? (N_SEQ - 1 - step) : step;
    const _Float16* pre_t = pre + (size_t)tt * B_SZ * H4;
    if (step + 1 < N_SEQ) {                   // pull next step's pre-activations toward L2/L0
      int tn = dir ? (N_SEQ - 2 - step) : (step + 1);
      __builtin_prefetch(pre + (size_t)tn * B_SZ * H4 + tid * 32, 0, 1);
    }
#pragma unroll
    for (int q = 0; q < 4; ++q) {
      int tile = w * 4 + q;                   // 0..127
      int mt = tile >> 6, nt = tile & 63;
      int m0 = mt * 16, n0 = nt * 16;
      v8f acc = {};
#pragma unroll
      for (int kc = 0; kc < H_DIM / 32; ++kc) {
        v16h a = frag_a(h_lds + (size_t)m0 * H_DIM + kc * 32, H_DIM);
        v16h f = frag_bt(Whh + (size_t)n0 * H_DIM + kc * 32, H_DIM);
        acc = wmma_f16(a, f, acc);
      }
      int cn = n0 + (l & 15), rb = (l >> 4) << 3;
#pragma unroll
      for (int r = 0; r < 8; ++r) {
        int mm = m0 + rb + r;
        z_lds[mm * H4 + cn] = acc[r] + (float)pre_t[(size_t)mm * H4 + cn];
      }
    }
    __syncthreads();
#pragma unroll
    for (int q = 0; q < 8; ++q) {
      int i = tid + q * 1024;                 // (b,u) flat, 8192 total
      int b = i >> 8, u = i & 255;
      float zi = z_lds[b * H4 + u];
      float zf = z_lds[b * H4 + 256 + u];
      float zg = z_lds[b * H4 + 512 + u];
      float zo = z_lds[b * H4 + 768 + u];
      float cNew = sigmoidf_(zf) * c_lds[i] + sigmoidf_(zi) * tanhf(zg);
      float hNew = sigmoidf_(zo) * tanhf(cNew);
      c_lds[i] = cNew;
      h_lds[i] = (_Float16)hNew;
      hout[(size_t)tt * B_SZ * H_DIM + i] = (_Float16)hNew;
    }
    __syncthreads();
  }
  if (dir == 0 && hT != nullptr) {
    for (int i = tid; i < B_SZ * H_DIM; i += 1024) {
      hT[i] = (float)h_lds[i];
      cT[i] = c_lds[i];
    }
  }
}

// -------------------- Th[i][b][h] = T @ concat(henc, hback)  (K = 512) --------------------
__global__ void __launch_bounds__(256)
k_th_gemm(const _Float16* __restrict__ henc, const _Float16* __restrict__ hback,
          const _Float16* __restrict__ Tm, _Float16* __restrict__ Th) {
  int wave = (blockIdx.x * blockDim.x + threadIdx.x) >> 5;
  int mt = wave >> 4;                 // 1024 M-tiles (N*B/16), m = i*32+b
  int nt = wave & 15;                 // 16 N-tiles (H/16)
  int m0 = mt * 16, n0 = nt * 16;
  v8f acc = {};
#pragma unroll
  for (int kc = 0; kc < 16; ++kc) {   // 512 / 32
    const _Float16* src = (kc < 8) ? henc : hback;
    int k0 = (kc < 8) ? kc * 32 : (kc - 8) * 32;
    v16h a = frag_a(src + (size_t)m0 * H_DIM + k0, H_DIM);
    v16h f = frag_bt(Tm + (size_t)n0 * (2 * H_DIM) + kc * 32, 2 * H_DIM);
    acc = wmma_f16(a, f, acc);
  }
  int l = lane_id(), cn = n0 + (l & 15), rb = (l >> 4) << 3;
#pragma unroll
  for (int r = 0; r < 8; ++r)
    Th[(size_t)(m0 + rb + r) * H_DIM + cn] = (_Float16)acc[r];
}

// -------------------- eij[b][i][j] = h_dec[b,j] . Th[b,i]  (per-batch 512x512x256) --------------------
// h_dec[b,j] = (j==0) ? henc[b, N-1] : dec_out[b, j-1]
__global__ void __launch_bounds__(256)
k_eij_gemm(const _Float16* __restrict__ Th, const _Float16* __restrict__ henc,
           const _Float16* __restrict__ dec_out, float* __restrict__ eij) {
  int wave = (blockIdx.x * blockDim.x + threadIdx.x) >> 5;
  int b = wave >> 10;
  int rem = wave & 1023;
  int it = rem >> 5, jt = rem & 31;
  int i0 = it * 16, j0 = jt * 16;
  int l = lane_id();
  v8f acc = {};
#pragma unroll
  for (int kc = 0; kc < H_DIM / 32; ++kc) {
    v16h a = frag_a(Th + ((size_t)i0 * 32 + b) * H_DIM + kc * 32, 32 * H_DIM);
    int n = j0 + (l & 15);
    int kb = kc * 32 + ((l >> 4) << 4);
    const _Float16* src = (n == 0)
        ? (henc + ((size_t)(N_SEQ - 1) * 32 + b) * H_DIM + kb)
        : (dec_out + ((size_t)(n - 1) * 32 + b) * H_DIM + kb);
    v16h f;
#pragma unroll
    for (int e = 0; e < 16; ++e) f[e] = src[e];
    acc = wmma_f16(a, f, acc);
  }
  int cn = j0 + (l & 15), rb = (l >> 4) << 3;
#pragma unroll
  for (int r = 0; r < 8; ++r)
    eij[((size_t)b * N_SEQ + (i0 + rb + r)) * N_SEQ + cn] = acc[r];
}

// -------------------- align softmax(i) * scores, log-sum, atomic accumulate --------------------
__global__ void __launch_bounds__(256)
k_final(const float* __restrict__ eij, const float* __restrict__ scores, float* __restrict__ out) {
  int wave = (blockIdx.x * blockDim.x + threadIdx.x) >> 5;
  int l = lane_id();
  int b = wave / N_SEQ, j = wave % N_SEQ;
  const float* ecol = eij    + (size_t)b * N_SEQ * N_SEQ + j;
  const float* scol = scores + (size_t)b * N_SEQ * N_SEQ + j;
  float mx = -1e30f;
  for (int i = l; i < N_SEQ; i += 32) mx = fmaxf(mx, ecol[(size_t)i * N_SEQ]);
#pragma unroll
  for (int s = 16; s > 0; s >>= 1) mx = fmaxf(mx, __shfl_xor(mx, s, 32));
  float Z = 0.f, acc = 0.f;
  for (int i = l; i < N_SEQ; i += 32) {
    float e = expf(ecol[(size_t)i * N_SEQ] - mx);
    Z += e;
    acc += e * scol[(size_t)i * N_SEQ];
  }
#pragma unroll
  for (int s = 16; s > 0; s >>= 1) { Z += __shfl_xor(Z, s, 32); acc += __shfl_xor(acc, s, 32); }
  if (l == 0) atomicAdd(out, -(logf(acc) - logf(Z)));
}

// ==================== host side ====================
extern "C" void kernel_launch(void* const* d_in, const int* in_sizes, int n_in,
                              void* d_out, int out_size, void* d_ws, size_t ws_size,
                              hipStream_t stream) {
  const int*   xs        = (const int*)d_in[0];
  const int*   ys        = (const int*)d_in[1];
  const float* gembed    = (const float*)d_in[2];
  const float* gconv     = (const float*)d_in[3];
  const float* gdecode   = (const float*)d_in[4];
  const float* enc_embed = (const float*)d_in[5];
  const float* dec_embed = (const float*)d_in[6];
  const float* enc_Wih   = (const float*)d_in[7];
  const float* enc_Whh   = (const float*)d_in[8];
  const float* enc_b     = (const float*)d_in[9];
  const float* dec_Wih   = (const float*)d_in[10];
  const float* dec_Whh   = (const float*)d_in[11];
  const float* dec_b     = (const float*)d_in[12];
  const float* Tmat      = (const float*)d_in[13];

  char* ws = (char*)d_ws;
  size_t off = 0;
  auto alloc = [&](size_t bytes) -> void* {
    void* p = ws + off;
    off = (off + bytes + 255) & ~(size_t)255;
    return p;
  };

  _Float16* gconvT_h  = (_Float16*)alloc((size_t)K_TAPS * F_DIM * F_DIM * 2);  // [k][fout][fin]
  _Float16* gdecT_h   = (_Float16*)alloc((size_t)OUT_V * F_DIM * 2);           // [v][fin]
  _Float16* encWih_h  = (_Float16*)alloc((size_t)H4 * E_DIM * 2);
  _Float16* encWhh_h  = (_Float16*)alloc((size_t)H4 * H_DIM * 2);
  _Float16* decWih_h  = (_Float16*)alloc((size_t)H4 * E_DIM * 2);
  _Float16* decWhh_h  = (_Float16*)alloc((size_t)H4 * H_DIM * 2);
  _Float16* T_h       = (_Float16*)alloc((size_t)H_DIM * 2 * H_DIM * 2);
  _Float16* epad      = (_Float16*)alloc((size_t)B_SZ * (N_SEQ + 4) * F_DIM * 2);
  _Float16* aconv     = (_Float16*)alloc((size_t)B_SZ * N_SEQ * F_DIM * 2);
  float*    scores    = (float*)alloc((size_t)B_SZ * N_SEQ * N_SEQ * 4);
  _Float16* pre_enc   = (_Float16*)alloc((size_t)N_SEQ * B_SZ * H4 * 2);
  _Float16* pre_dec   = (_Float16*)alloc((size_t)N_SEQ * B_SZ * H4 * 2);
  _Float16* henc      = (_Float16*)alloc((size_t)N_SEQ * B_SZ * H_DIM * 2);
  _Float16* hback     = (_Float16*)alloc((size_t)N_SEQ * B_SZ * H_DIM * 2);
  _Float16* dout      = (_Float16*)alloc((size_t)N_SEQ * B_SZ * H_DIM * 2);
  float*    hT        = (float*)alloc((size_t)B_SZ * H_DIM * 4);
  float*    cT        = (float*)alloc((size_t)B_SZ * H_DIM * 4);
  _Float16* Th        = (_Float16*)alloc((size_t)N_SEQ * B_SZ * H_DIM * 2);
  float*    eij       = (float*)alloc((size_t)B_SZ * N_SEQ * N_SEQ * 4);

  const size_t sm_decode = 16 * F_DIM * sizeof(_Float16) + 16 * OUT_V * sizeof(float);
  const size_t sm_lstm   = (size_t)B_SZ * H_DIM * sizeof(_Float16)
                         + (size_t)B_SZ * H_DIM * sizeof(float)
                         + (size_t)B_SZ * H4 * sizeof(float);
  (void)hipFuncSetAttribute((const void*)k_decode_softmax_gather,
                            hipFuncAttributeMaxDynamicSharedMemorySize, (int)sm_decode);
  (void)hipFuncSetAttribute((const void*)k_lstm,
                            hipFuncAttributeMaxDynamicSharedMemorySize, (int)sm_lstm);

  k_init_out<<<1, 64, 0, stream>>>((float*)d_out);

  auto cvt = [&](const float* s, _Float16* d, int n) {
    k_f32_to_f16<<<(n + 255) / 256, 256, 0, stream>>>(s, d, n);
  };
  auto cvtT = [&](const float* s, _Float16* d, int R, int C, int bat) {
    int n = bat * R * C;
    k_transpose_to_f16<<<(n + 255) / 256, 256, 0, stream>>>(s, d, R, C, n);
  };
  cvtT(gconv,   gconvT_h, F_DIM, F_DIM, K_TAPS);  // [k][fin][fout] -> [k][fout][fin]
  cvtT(gdecode, gdecT_h,  F_DIM, OUT_V, 1);       // [fin][v] -> [v][fin]
  cvt(enc_Wih, encWih_h,  H4 * E_DIM);            // already N-major for B^T use
  cvt(enc_Whh, encWhh_h,  H4 * H_DIM);
  cvt(dec_Wih, decWih_h,  H4 * E_DIM);
  cvt(dec_Whh, decWhh_h,  H4 * H_DIM);
  cvt(Tmat,    T_h,       H_DIM * 2 * H_DIM);

  {
    int n = B_SZ * (N_SEQ + 4) * F_DIM;
    k_embed_tanh_pad<<<(n + 255) / 256, 256, 0, stream>>>(xs, gembed, epad);
  }
  k_conv_gemm<<<2048, 256, 0, stream>>>(epad, gconvT_h, aconv);                  // 16384 waves
  k_decode_softmax_gather<<<B_SZ * N_SEQ / 16, 256, sm_decode, stream>>>(aconv, gdecT_h, ys, scores);
  k_pre_gemm<<<8192, 256, 0, stream>>>(xs, enc_embed, encWih_h, enc_b, pre_enc); // 65536 waves
  k_pre_gemm<<<8192, 256, 0, stream>>>(ys, dec_embed, decWih_h, dec_b, pre_dec);
  // encoder fwd + bwd in parallel (two WGPs), then decoder seeded by (hT,cT)
  k_lstm<<<2, 1024, sm_lstm, stream>>>(pre_enc, encWhh_h, henc, hback, nullptr, nullptr, hT, cT);
  k_lstm<<<1, 1024, sm_lstm, stream>>>(pre_dec, decWhh_h, dout, dout, hT, cT, nullptr, nullptr);
  k_th_gemm<<<2048, 256, 0, stream>>>(henc, hback, T_h, Th);                     // 16384 waves
  k_eij_gemm<<<4096, 256, 0, stream>>>(Th, henc, dout, eij);                     // 32768 waves
  k_final<<<2048, 256, 0, stream>>>(eij, scores, (float*)d_out);                 // wave per (b,j)
}
